// LSTM_87557203296680
// MI455X (gfx1250) — matrix operand order repaired and checked by reading, not compiled, not run on hardware
//
#include <hip/hip_runtime.h>
#include <hip/hip_bf16.h>
#include <math.h>

// ---------------------------------------------------------------------------
// 2-layer LSTM (B=256, T=512, IN=8, H=256) + FC head, CDNA5 / gfx1250.
// Batch-split persistent workgroups (16 WGs x 16 batch rows), recurrent state
// in LDS, v_wmma_f32_16x16x32_bf16, input projection fused via K-concat.
// A-operand staged in LDS in contiguous k order (same as B); weight
// tiles streamed from L2 with a half-depth (4-tile) register ping-pong so 8
// loads are always in flight; layer-1 inputs staged via async global->LDS DMA.
// ---------------------------------------------------------------------------

typedef __attribute__((ext_vector_type(16))) __bf16 v16bf;
typedef __attribute__((ext_vector_type(8)))  __bf16 v8bf;
typedef __attribute__((ext_vector_type(8)))  float  v8f;
typedef __attribute__((ext_vector_type(4)))  int    v4i;

#define LSTM_B   256
#define LSTM_T   512
#define LSTM_IN  8
#define LSTM_H   256
#define LSTM_G   1024   // 4*H
#define FC_HZ    2
#define MTILE    16     // batch rows per workgroup
#define THREADS  256    // 8 waves (wave32)

#if __has_builtin(__builtin_amdgcn_global_load_async_to_lds_b128)
#define HAVE_ASYNC_LDS 1
#endif

__device__ __forceinline__ void wait_async_all() {
#if defined(HAVE_ASYNC_LDS)
#if __has_builtin(__builtin_amdgcn_s_wait_asynccnt)
    __builtin_amdgcn_s_wait_asynccnt(0);
#else
    asm volatile("s_wait_asynccnt 0x0" ::: "memory");
#endif
#endif
}

__device__ __forceinline__ float sigmoidf_(float x) {
    return 1.0f / (1.0f + __expf(-x));
}

// A-fragment layout within a 32-wide K group (identity: contiguous k order).
__device__ __forceinline__ int kswz(int s) { return s; }

// --------------------------- weight prepacking -----------------------------
// Layer 0: Kcat=288, x part k[0..7], zero pad k[8..31], W_hh at k[32..287]
__global__ void pack_l0(const float* __restrict__ Wih, const float* __restrict__ Whh,
                        const float* __restrict__ bih, const float* __restrict__ bhh,
                        __bf16* __restrict__ Wcat, float* __restrict__ bias) {
    int n = blockIdx.x * blockDim.x + threadIdx.x;
    if (n >= LSTM_G) return;
    __bf16* row = Wcat + (size_t)n * 288;
    for (int k = 0; k < LSTM_IN; ++k)  row[k]      = (__bf16)Wih[n * LSTM_IN + k];
    for (int k = LSTM_IN; k < 32; ++k) row[k]      = (__bf16)0.0f;
    for (int k = 0; k < LSTM_H; ++k)   row[32 + k] = (__bf16)Whh[n * LSTM_H + k];
    bias[n] = bih[n] + bhh[n];
}

// Layer 1: Kcat=512, [ W_ih1 | W_hh1 ]
__global__ void pack_l1(const float* __restrict__ Wih, const float* __restrict__ Whh,
                        const float* __restrict__ bih, const float* __restrict__ bhh,
                        __bf16* __restrict__ Wcat, float* __restrict__ bias) {
    int n = blockIdx.x * blockDim.x + threadIdx.x;
    if (n >= LSTM_G) return;
    __bf16* row = Wcat + (size_t)n * 512;
    for (int k = 0; k < LSTM_H; ++k) {
        row[k]          = (__bf16)Wih[n * LSTM_H + k];
        row[LSTM_H + k] = (__bf16)Whh[n * LSTM_H + k];
    }
    bias[n] = bih[n] + bhh[n];
}

// --------------------------- recurrent layer -------------------------------
template <bool IS_L0, int KCAT, int KX>
__global__ void lstm_layer(const void* __restrict__ xseq_v,
                           const __bf16* __restrict__ Wcat,
                           const float* __restrict__ bias,
                           __bf16* __restrict__ hseq_out,   // L0: [B][T][H] bf16
                           float*  __restrict__ hlast) {    // L1: [B][H] f32 (t = T-1)
    constexpr int KSTEPS = KCAT / 32;

    extern __shared__ char smem[];
    float*  gates = (float*)smem;                       // 16*1024 f32 = 64 KB
    float*  cst   = gates + MTILE * LSTM_G;             // 16*256  f32 = 16 KB
    float*  bLds  = cst   + MTILE * LSTM_H;             // 1024    f32 =  4 KB
    __bf16* Abuf  = (__bf16*)(bLds + LSTM_G);           // 16*KCAT bf16 (32B aligned)

    const int tid = threadIdx.x;
    const int b0  = blockIdx.x * MTILE;

    for (int i = tid; i < MTILE * KCAT;   i += THREADS) Abuf[i] = (__bf16)0.0f;
    for (int i = tid; i < MTILE * LSTM_H; i += THREADS) cst[i]  = 0.0f;
    for (int i = tid; i < LSTM_G;         i += THREADS) bLds[i] = bias[i];
    __syncthreads();

    const int wave = tid >> 5;
    const int lane = tid & 31;
    const int lr   = lane & 15;
    const int hi   = lane >> 4;

    const __bf16* abase = Abuf + lr * KCAT + hi * 16;
    const __bf16* wbase = Wcat + (size_t)(wave * 128 + lr) * KCAT + hi * 16;

    // staging geometry for layer-1 inputs (one 16-element chunk per thread)
    const int srow  = tid >> 4;
    const int soff  = (tid & 15) * 16;
    const int sg32  = soff & ~31;
    const int shalf = (soff >> 4) & 1;
    __bf16* sdst = Abuf + srow * KCAT + sg32 + shalf * 16;  // contiguous dest

    // hoist bias to registers (constant over time)
    float bv[8];
#pragma unroll
    for (int nt = 0; nt < 8; ++nt) bv[nt] = bLds[(wave * 8 + nt) * 16 + lr];

    for (int t = 0; t < LSTM_T; ++t) {
        // ---- stage x_t into the front of Acat (contiguous k order) ----
        if (IS_L0) {
            const float* x = (const float*)xseq_v;
            if (tid < MTILE * LSTM_IN) {
                int row = tid >> 3, col = tid & 7;      // k%32 < 8 -> identity swizzle
                const float* xp = x + ((size_t)(b0 + row) * LSTM_T + t) * LSTM_IN + col;
                Abuf[row * KCAT + col] = (__bf16)(*xp);
                if (t + 1 < LSTM_T) __builtin_prefetch(xp + LSTM_IN, 0, 1);
            }
        } else {
            const __bf16* x = (const __bf16*)xseq_v;
            const __bf16* src = x + ((size_t)(b0 + srow) * LSTM_T + t) * LSTM_H + soff;
#if defined(HAVE_ASYNC_LDS)
            if (t == 0) {   // first step staged synchronously
                *(v8bf*)sdst        = *(const v8bf*)src;
                *(v8bf*)(sdst + 8) = *(const v8bf*)(src + 8);
            } else {
                wait_async_all();   // x_t DMA (issued at t-1) complete
            }
#else
            *(v8bf*)sdst        = *(const v8bf*)src;
            *(v8bf*)(sdst + 8) = *(const v8bf*)(src + 8);
#endif
        }
        __syncthreads();   // Acat (x and h) fully written before fragment loads

        // ---- gates tile: 8 N-tiles per wave, half-depth (4-tile) ping-pong ----
        v8f acc[8];
#pragma unroll
        for (int nt = 0; nt < 8; ++nt)
#pragma unroll
            for (int j = 0; j < 8; ++j) acc[nt][j] = bv[nt];

        v16bf bLo[4], bHi[4];
#pragma unroll
        for (int nt = 0; nt < 4; ++nt)
            bLo[nt] = *(const v16bf*)(wbase + (size_t)nt * 16 * KCAT);

#pragma unroll 1
        for (int k = 0; k < KSTEPS; ++k) {
            // loads for tiles 4..7 of this k (consumed in phase B)
#pragma unroll
            for (int nt = 0; nt < 4; ++nt)
                bHi[nt] = *(const v16bf*)(wbase + (size_t)(nt + 4) * 16 * KCAT + k * 32);

            const v16bf av = *(const v16bf*)(abase + k * 32);   // one 32B LDS read
#pragma unroll
            for (int nt = 0; nt < 4; ++nt)
                acc[nt] = __builtin_amdgcn_wmma_f32_16x16x32_bf16(
                              false, av, false, bLo[nt], (short)0, acc[nt], false, false);

            // refill tiles 0..3 for k+1 (clamped dummy reload on last iter)
            const int kn = (k + 1 < KSTEPS) ? k + 1 : k;
#pragma unroll
            for (int nt = 0; nt < 4; ++nt)
                bLo[nt] = *(const v16bf*)(wbase + (size_t)nt * 16 * KCAT + kn * 32);

#pragma unroll
            for (int nt = 0; nt < 4; ++nt)
                acc[nt + 4] = __builtin_amdgcn_wmma_f32_16x16x32_bf16(
                                  false, av, false, bHi[nt], (short)0, acc[nt + 4], false, false);
        }

#pragma unroll
        for (int nt = 0; nt < 8; ++nt)
#pragma unroll
            for (int j = 0; j < 8; ++j)
                gates[(hi * 8 + j) * LSTM_G + (wave * 8 + nt) * 16 + lr] = acc[nt][j];
        __syncthreads();   // gates complete, all A-fragment reads of step t done

        // ---- issue async DMA of x_{t+1} into the (now idle) A x-region ----
#if defined(HAVE_ASYNC_LDS)
        if (!IS_L0 && t + 1 < LSTM_T) {
            const __bf16* x = (const __bf16*)xseq_v;
            const __bf16* nsrc =
                x + ((size_t)(b0 + srow) * LSTM_T + (t + 1)) * LSTM_H + soff;
            __builtin_amdgcn_global_load_async_to_lds_b128(
                (v4i*)nsrc, (v4i*)sdst, 0, 0);
            __builtin_amdgcn_global_load_async_to_lds_b128(
                (v4i*)(nsrc + 8), (v4i*)(sdst + 8), 0, 0);
        }
#endif

        // ---- activations: thread handles 16 hidden units of one row ----
        {
            int row = tid & 15;
            int ub  = (tid >> 4) * 16;
            const float* g = gates + row * LSTM_G;
            float*  cp = cst + row * LSTM_H;
            __bf16* hp = Abuf + row * KCAT + KX;    // KX % 32 == 0
#pragma unroll 4
            for (int s = 0; s < 16; ++s) {
                int u = ub + s;
                float iv = sigmoidf_(g[u]);
                float fv = sigmoidf_(g[256 + u]);
                float gv = tanhf(g[512 + u]);
                float ov = sigmoidf_(g[768 + u]);
                float cn = fv * cp[u] + iv * gv;
                cp[u] = cn;
                float hv = ov * tanhf(cn);
                hp[(u & ~31) + kswz(u & 31)] = (__bf16)hv;
                if (IS_L0) {
                    hseq_out[((size_t)(b0 + row) * LSTM_T + t) * LSTM_H + u] = (__bf16)hv;
                } else if (t == LSTM_T - 1) {
                    hlast[(size_t)(b0 + row) * LSTM_H + u] = hv;
                }
            }
        }
        // next iteration's leading __syncthreads orders h writes vs. A reads
    }
}

// ------------------------------- FC head -----------------------------------
__global__ void fc_kernel(const float* __restrict__ hlast,
                          const float* __restrict__ fcw,
                          const float* __restrict__ fcb,
                          float* __restrict__ out) {
    int idx = blockIdx.x * blockDim.x + threadIdx.x;
    if (idx >= LSTM_B * FC_HZ) return;
    int b = idx >> 1, hz = idx & 1;
    const float* h = hlast + (size_t)b * LSTM_H;
    const float* w = fcw + (size_t)hz * LSTM_H;
    float acc = fcb[hz];
#pragma unroll 8
    for (int k = 0; k < LSTM_H; ++k) acc = fmaf(h[k], w[k], acc);
    out[idx] = acc;
}

// ------------------------------- launcher ----------------------------------
extern "C" void kernel_launch(void* const* d_in, const int* in_sizes, int n_in,
                              void* d_out, int out_size, void* d_ws, size_t ws_size,
                              hipStream_t stream) {
    const float* x    = (const float*)d_in[0];
    const float* Wih0 = (const float*)d_in[1];
    const float* Whh0 = (const float*)d_in[2];
    const float* bih0 = (const float*)d_in[3];
    const float* bhh0 = (const float*)d_in[4];
    const float* Wih1 = (const float*)d_in[5];
    const float* Whh1 = (const float*)d_in[6];
    const float* bih1 = (const float*)d_in[7];
    const float* bhh1 = (const float*)d_in[8];
    const float* fcw  = (const float*)d_in[9];
    const float* fcb  = (const float*)d_in[10];

    char* ws = (char*)d_ws;
    __bf16* Wcat0 = (__bf16*)(ws + 0);          // 1024*288*2 = 589824
    float*  bias0 = (float*)(ws + 589824);      // 4096
    __bf16* Wcat1 = (__bf16*)(ws + 593920);     // 1048576
    float*  bias1 = (float*)(ws + 1642496);     // 4096
    __bf16* h0seq = (__bf16*)(ws + 1646592);    // 256*512*256*2 = 67108864
    float*  hlast = (float*)(ws + 68755456);    // 256*256*4 = 262144

    pack_l0<<<dim3(4), dim3(256), 0, stream>>>(Wih0, Whh0, bih0, bhh0, Wcat0, bias0);
    pack_l1<<<dim3(4), dim3(256), 0, stream>>>(Wih1, Whh1, bih1, bhh1, Wcat1, bias1);

    const size_t shBase = (size_t)(MTILE * LSTM_G + MTILE * LSTM_H + LSTM_G) * 4;
    const size_t sh0 = shBase + (size_t)MTILE * 288 * 2;   // ~93 KB
    const size_t sh1 = shBase + (size_t)MTILE * 512 * 2;   // ~100 KB

    lstm_layer<true, 288, 32>
        <<<dim3(LSTM_B / MTILE), dim3(THREADS), sh0, stream>>>(
            (const void*)x, Wcat0, bias0, h0seq, nullptr);

    lstm_layer<false, 512, 256>
        <<<dim3(LSTM_B / MTILE), dim3(THREADS), sh1, stream>>>(
            (const void*)h0seq, Wcat1, bias1, nullptr, hlast);

    fc_kernel<<<dim3(2), dim3(256), 0, stream>>>(hlast, fcw, fcb, (float*)d_out);
}
